// Point_M2AE_ReID_41661182771613
// MI455X (gfx1250) — matrix operand order repaired and checked by reading, not compiled, not run on hardware
//
#include <hip/hip_runtime.h>
#include <cmath>

// ---------------------------------------------------------------------------
// Types for CDNA5 WMMA (wave32, 16x16x32 bf16 -> f32)
// ---------------------------------------------------------------------------
typedef __bf16 bf16_t;
typedef bf16_t v16bf __attribute__((ext_vector_type(16)));
typedef bf16_t v8bf  __attribute__((ext_vector_type(8)));
typedef bf16_t v4bf  __attribute__((ext_vector_type(4)));
typedef float  v8f   __attribute__((ext_vector_type(8)));

__device__ __forceinline__ float gelu_exact(float x) {
    return 0.5f * x * (1.0f + erff(x * 0.70710678118654752f));
}

// ---------------------------------------------------------------------------
// Generic WMMA GEMM:
//   C[m,n] = EPI( alpha * sum_k A[m,k]*B(k,n) + bias[n] )
//   B(k,n) = TRANSB ? Bm[n*ldb + k] : Bm[k*ldb + n]
// blockIdx.z -> (zb = z/H, zh = z%H); pointers offset by zb*bs + zh*hs.
// EPI: 0 = none, 1 = relu, 2 = gelu(exact), 3 = BN(g,b,m,v)+relu
//
// Block tile 128x64, 256 threads = 8 waves in 4(m) x 2(n); each wave owns a
// 2x2 grid of 16x16 WMMA tiles (4 v_wmma per K-step, A/B fragment reuse).
// K stepped by 32. A staged row-major [128][40] bf16; B staged TRANSPOSED
// [n][k] = [64][40] bf16 so fragments are two aligned 16B ds loads each.
// Interior tiles take a uniform fast path: float4 global loads + v4bf (8B)
// LDS stores; edge/misaligned tiles take the guarded scalar path.
// ---------------------------------------------------------------------------
template <int EPI, int TRANSB>
__global__ void __launch_bounds__(256)
gemm_wmma(const float* __restrict__ A, int lda, long bsA, long hsA,
          const float* __restrict__ Bm, int ldb, long bsB, long hsB,
          const float* __restrict__ bias,
          const float* __restrict__ bn_g, const float* __restrict__ bn_b,
          const float* __restrict__ bn_m, const float* __restrict__ bn_v,
          float* __restrict__ C, int ldc, long bsC, long hsC,
          int M, int N, int K, int H, float alpha)
{
    const int zb = blockIdx.z / H, zh = blockIdx.z % H;
    A  += (long)zb * bsA + (long)zh * hsA;
    Bm += (long)zb * bsB + (long)zh * hsB;
    C  += (long)zb * bsC + (long)zh * hsC;

    __shared__ bf16_t As[128][40];   // 80B row pitch (16B aligned)
    __shared__ bf16_t Bst[64][40];   // transposed: [n][k]

    const int tid  = threadIdx.x;
    const int lane = tid & 31;
    const int w    = tid >> 5;       // 0..7
    const int wm   = w & 3;          // 4 wave-tiles along M (32 rows each)
    const int wn   = w >> 2;         // 2 wave-tiles along N (32 cols each)
    const int tileM = blockIdx.x * 128;
    const int tileN = blockIdx.y * 64;

    const bool vecA = ((lda & 3) == 0) && ((((unsigned long)A)  & 15) == 0);
    const bool vecB = TRANSB && ((ldb & 3) == 0) && ((((unsigned long)Bm) & 15) == 0);
    const bool fullM = (tileM + 128 <= M);
    const bool fullN = (tileN + 64 <= N);

    v8f acc00 = {}, acc01 = {}, acc10 = {}, acc11 = {};

    for (int k0 = 0; k0 < K; k0 += 32) {
        const bool fullK = (k0 + 32 <= K);
        if (k0 + 32 < K)  // hint next K-tile of A (-> global_prefetch_b8)
            __builtin_prefetch(A + (long)tileM * lda + (k0 + 32), 0, 1);

        // ---- stage A: 128x32, four 4-element chunks per thread ----
        if (vecA && fullM && fullK) {
            #pragma unroll
            for (int t = 0; t < 4; ++t) {
                int chunk = tid + t * 256;          // 0..1023
                int r = chunk >> 3, c4 = (chunk & 7) << 2;
                const float4 vv = *(const float4*)(A + (long)(tileM + r) * lda + (k0 + c4));
                v4bf pk = { (bf16_t)vv.x, (bf16_t)vv.y, (bf16_t)vv.z, (bf16_t)vv.w };
                *(v4bf*)&As[r][c4] = pk;
            }
        } else {
            #pragma unroll
            for (int t = 0; t < 4; ++t) {
                int chunk = tid + t * 256;
                int r = chunk >> 3, c4 = (chunk & 7) << 2;
                int gm = tileM + r;
                v4bf pk;
                #pragma unroll
                for (int u = 0; u < 4; ++u) {
                    int gk = k0 + c4 + u;
                    float v = (gm < M && gk < K) ? A[(long)gm * lda + gk] : 0.0f;
                    pk[u] = (bf16_t)v;
                }
                *(v4bf*)&As[r][c4] = pk;
            }
        }

        // ---- stage B transposed: Bst[n][k], 64x32, two chunks per thread ----
        if (vecB && fullN && fullK) {
            #pragma unroll
            for (int t = 0; t < 2; ++t) {
                int chunk = tid + t * 256;          // 0..511
                int n = chunk >> 3, k4 = (chunk & 7) << 2;
                const float4 vv = *(const float4*)(Bm + (long)(tileN + n) * ldb + (k0 + k4));
                v4bf pk = { (bf16_t)vv.x, (bf16_t)vv.y, (bf16_t)vv.z, (bf16_t)vv.w };
                *(v4bf*)&Bst[n][k4] = pk;
            }
        } else {
            #pragma unroll
            for (int t = 0; t < 2; ++t) {
                int chunk = tid + t * 256;
                int n = chunk >> 3, k4 = (chunk & 7) << 2;
                int gn = tileN + n;
                v4bf pk;
                #pragma unroll
                for (int u = 0; u < 4; ++u) {
                    int gk = k0 + k4 + u;
                    float v = 0.0f;
                    if (gn < N && gk < K)
                        v = TRANSB ? Bm[(long)gn * ldb + gk] : Bm[(long)gk * ldb + gn];
                    pk[u] = (bf16_t)v;
                }
                *(v4bf*)&Bst[n][k4] = pk;
            }
        }
        __syncthreads();

        // ---- build fragments (all aligned 16B LDS loads) ----
        // A 16-bit 16x32 layout: lanes 0-15 row m hold K {0..7,16..23};
        // lanes 16-31 row m hold K {8..15,24..31}.
        const int m     = lane & 15;
        const int abase = (lane < 16) ? 0 : 8;
        const int koff  = (lane < 16) ? 0 : 16;

        v16bf a0, a1, b0, b1;
        {
            const int r0 = wm * 32 + m;
            v8bf lo = *(const v8bf*)&As[r0][abase];
            v8bf hi = *(const v8bf*)&As[r0][16 + abase];
            a0 = __builtin_shufflevector(lo, hi, 0,1,2,3,4,5,6,7,8,9,10,11,12,13,14,15);
            const int r1 = r0 + 16;
            lo = *(const v8bf*)&As[r1][abase];
            hi = *(const v8bf*)&As[r1][16 + abase];
            a1 = __builtin_shufflevector(lo, hi, 0,1,2,3,4,5,6,7,8,9,10,11,12,13,14,15);
        }
        {
            const int c0 = wn * 32 + m;
            v8bf lo = *(const v8bf*)&Bst[c0][abase];
            v8bf hi = *(const v8bf*)&Bst[c0][16 + abase];
            b0 = __builtin_shufflevector(lo, hi, 0,1,2,3,4,5,6,7,8,9,10,11,12,13,14,15);
            const int c1 = c0 + 16;
            lo = *(const v8bf*)&Bst[c1][abase];
            hi = *(const v8bf*)&Bst[c1][16 + abase];
            b1 = __builtin_shufflevector(lo, hi, 0,1,2,3,4,5,6,7,8,9,10,11,12,13,14,15);
        }

        acc00 = __builtin_amdgcn_wmma_f32_16x16x32_bf16(false, a0, false, b0, (short)0, acc00, false, false);
        acc01 = __builtin_amdgcn_wmma_f32_16x16x32_bf16(false, a0, false, b1, (short)0, acc01, false, false);
        acc10 = __builtin_amdgcn_wmma_f32_16x16x32_bf16(false, a1, false, b0, (short)0, acc10, false, false);
        acc11 = __builtin_amdgcn_wmma_f32_16x16x32_bf16(false, a1, false, b1, (short)0, acc11, false, false);
        __syncthreads();
    }

    // ---- epilogue. C/D layout: lane col = lane%16; rows r (lanes<16) / 8+r.
    const bool hasBias = (bias != nullptr);
    #pragma unroll
    for (int sm = 0; sm < 2; ++sm) {
        #pragma unroll
        for (int sn = 0; sn < 2; ++sn) {
            const v8f& acc = (sm == 0) ? (sn == 0 ? acc00 : acc01)
                                       : (sn == 0 ? acc10 : acc11);
            const int gn = tileN + wn * 32 + sn * 16 + (lane & 15);
            if (gn >= N) continue;
            float bv = hasBias ? bias[gn] : 0.0f;
            float bg = 0.f, bb2 = 0.f, bm2 = 0.f, bvv = 1.f;
            if (EPI == 3) { bg = bn_g[gn]; bb2 = bn_b[gn]; bm2 = bn_m[gn]; bvv = bn_v[gn]; }
            #pragma unroll
            for (int r = 0; r < 8; ++r) {
                const int gm = tileM + wm * 32 + sm * 16 + ((lane < 16) ? r : 8 + r);
                if (gm < M) {
                    float v = acc[r] * alpha + bv;
                    if (EPI == 1) v = fmaxf(v, 0.0f);
                    else if (EPI == 2) v = gelu_exact(v);
                    else if (EPI == 3) {
                        v = (v - bm2) * rsqrtf(bvv + 1e-5f) * bg + bb2;
                        v = fmaxf(v, 0.0f);
                    }
                    C[(long)gm * ldc + gn] = v;
                }
            }
        }
    }
}

// ---------------------------------------------------------------------------
// Farthest point sampling: one block per batch. Points are staged into LDS
// with the CDNA5 async global->LDS engine (ASYNCcnt path): fire-and-forget
// global_load_async_to_lds_b128 per 16B chunk, then s_wait_asynccnt 0.
// Low 32 bits of a generic LDS pointer are the wave-relative LDS offset
// (ISA 10.2: aperture lives in addr[63:32]), so they form the VDST operand.
// ---------------------------------------------------------------------------
__global__ void __launch_bounds__(256)
fps_kernel(const float* __restrict__ src, int N, int nsamp, int* __restrict__ out)
{
    const int b = blockIdx.x;
    __shared__ __attribute__((aligned(16))) float spts[2048 * 3];  // xyz interleaved
    __shared__ float dist[2048];
    __shared__ float rv[256];
    __shared__ int   ri[256];
    __shared__ int   s_last;

    const float* p = src + (long)b * N * 3;

    // ---- async stage: N*3 floats, 16B per transfer (N*3 divisible by 4) ----
    const int nchunks = (N * 3) >> 2;
    for (int c = threadIdx.x; c < nchunks; c += 256) {
        const float* gaddr = p + (c << 2);
        unsigned ldsoff = (unsigned)(unsigned long)&spts[c << 2];
        asm volatile("global_load_async_to_lds_b128 %0, %1, off"
                     :: "v"(ldsoff), "v"(gaddr) : "memory");
    }
    asm volatile("s_wait_asynccnt 0x0" ::: "memory");

    for (int i = threadIdx.x; i < N; i += 256) dist[i] = 1e10f;
    if (threadIdx.x == 0) { out[(long)b * nsamp] = 0; s_last = 0; }
    __syncthreads();

    for (int s = 1; s < nsamp; ++s) {
        const int last = s_last;
        const float lx = spts[last * 3 + 0];
        const float ly = spts[last * 3 + 1];
        const float lz = spts[last * 3 + 2];
        float bestv = -1.0f; int besti = N;
        for (int i = threadIdx.x; i < N; i += 256) {
            float dx = spts[i * 3 + 0] - lx;
            float dy = spts[i * 3 + 1] - ly;
            float dz = spts[i * 3 + 2] - lz;
            float d  = dx * dx + dy * dy + dz * dz;
            float dm = fminf(dist[i], d);
            dist[i] = dm;
            if (dm > bestv) { bestv = dm; besti = i; }
        }
        rv[threadIdx.x] = bestv; ri[threadIdx.x] = besti;
        __syncthreads();
        for (int off = 128; off > 0; off >>= 1) {
            if (threadIdx.x < off) {
                float v2 = rv[threadIdx.x + off]; int i2 = ri[threadIdx.x + off];
                if (v2 > rv[threadIdx.x] ||
                    (v2 == rv[threadIdx.x] && i2 < ri[threadIdx.x])) {
                    rv[threadIdx.x] = v2; ri[threadIdx.x] = i2;
                }
            }
            __syncthreads();
        }
        if (threadIdx.x == 0) { s_last = ri[0]; out[(long)b * nsamp + s] = ri[0]; }
        __syncthreads();
    }
}

// ---------------------------------------------------------------------------
// KNN: one thread per (b,g); register insertion-sort top-k (ascending d2).
// ---------------------------------------------------------------------------
template <int KN>
__global__ void __launch_bounds__(128)
knn_kernel(const float* __restrict__ pts, const float* __restrict__ ctr,
           int N, int G, int BG, int* __restrict__ out)
{
    const int t = blockIdx.x * 128 + threadIdx.x;
    if (t >= BG) return;
    const int b = t / G;
    const float* p = pts + (long)b * N * 3;
    const float cx = ctr[(long)t * 3 + 0];
    const float cy = ctr[(long)t * 3 + 1];
    const float cz = ctr[(long)t * 3 + 2];
    float bd[KN]; int bi[KN];
    #pragma unroll
    for (int j = 0; j < KN; ++j) { bd[j] = 1e30f; bi[j] = 0; }
    for (int i = 0; i < N; ++i) {
        float dx = p[i * 3] - cx, dy = p[i * 3 + 1] - cy, dz = p[i * 3 + 2] - cz;
        float d = dx * dx + dy * dy + dz * dz;
        if (d < bd[KN - 1]) {
            int j = KN - 1;
            while (j > 0 && bd[j - 1] > d) { bd[j] = bd[j - 1]; bi[j] = bi[j - 1]; --j; }
            bd[j] = d; bi[j] = i;
        }
    }
    #pragma unroll
    for (int j = 0; j < KN; ++j) out[(long)t * KN + j] = bi[j];
}

// dst[b,g,c] = src[b, idx[b,g], c]
__global__ void gather_points(const float* __restrict__ src, const int* __restrict__ idx,
                              float* __restrict__ dst, int Nsrc, int G, int C, long total)
{
    long t = (long)blockIdx.x * blockDim.x + threadIdx.x;
    if (t >= total) return;
    int c = (int)(t % C); long r = t / C;
    int g = (int)(r % G); int b = (int)(r / G);
    dst[t] = src[((long)b * Nsrc + idx[(long)b * G + g]) * C + c];
}

// dst[b,g,k,c] = pts[b, idx[b,g,k], c] - ctr[b,g,c]   (c in 0..2)
__global__ void gather_sub(const float* __restrict__ pts, const int* __restrict__ idx,
                           const float* __restrict__ ctr, float* __restrict__ dst,
                           int N, int G, int K, long total)
{
    long t = (long)blockIdx.x * blockDim.x + threadIdx.x;
    if (t >= total) return;
    int c = (int)(t % 3); long r = t / 3;
    int k = (int)(r % K); r /= K;
    int g = (int)(r % G); int b = (int)(r / G);
    int s = idx[((long)b * G + g) * K + k];
    dst[t] = pts[((long)b * N + s) * 3 + c] - ctr[((long)b * G + g) * 3 + c];
}

// dst[b,g,k,c] = x[b, idx[b,g,k], c] ; x: (B,Gp,Cp)
__global__ void gather_feats(const float* __restrict__ x, const int* __restrict__ idx,
                             float* __restrict__ dst, int Gp, int Cp, int G, int K, long total)
{
    long t = (long)blockIdx.x * blockDim.x + threadIdx.x;
    if (t >= total) return;
    int c = (int)(t % Cp); long r = t / Cp;
    int k = (int)(r % K); r /= K;
    int g = (int)(r % G); int b = (int)(r / G);
    int s = idx[((long)b * G + g) * K + k];
    dst[t] = x[((long)b * Gp + s) * Cp + c];
}

// y = a + b
__global__ void add_kernel(const float* __restrict__ a, const float* __restrict__ b,
                           float* __restrict__ y, long n)
{
    long t = (long)blockIdx.x * blockDim.x + threadIdx.x;
    if (t < n) y[t] = a[t] + b[t];
}

// f: (R,K,h) -> out: (R,K,2h) = [broadcast(max_k f), f]
__global__ void __launch_bounds__(128)
concat_maxpool(const float* __restrict__ f, float* __restrict__ out, int K, int h)
{
    const long r = blockIdx.x;
    const float* fr = f + r * (long)K * h;
    float* orow = out + r * (long)K * 2 * h;
    for (int c = threadIdx.x; c < h; c += 128) {
        float mx = -1e30f;
        for (int k = 0; k < K; ++k) mx = fmaxf(mx, fr[(long)k * h + c]);
        for (int k = 0; k < K; ++k) {
            orow[(long)k * 2 * h + c]     = mx;
            orow[(long)k * 2 * h + h + c] = fr[(long)k * h + c];
        }
    }
}

// out[r,c] = max_k f[(r*K+k), c]
__global__ void maxpool_k(const float* __restrict__ f, float* __restrict__ out,
                          int K, int D, long total)
{
    long t = (long)blockIdx.x * blockDim.x + threadIdx.x;
    if (t >= total) return;
    int c = (int)(t % D); long r = t / D;
    const float* fr = f + (r * K) * (long)D + c;
    float m = -1e30f;
    for (int k = 0; k < K; ++k) m = fmaxf(m, fr[(long)k * D]);
    out[t] = m;
}

// row-wise LayerNorm, one block (128 thr) per row
__global__ void __launch_bounds__(128)
layernorm_kernel(const float* __restrict__ x, const float* __restrict__ g,
                 const float* __restrict__ b, float* __restrict__ y, int D)
{
    const long r = blockIdx.x;
    __shared__ float red[128];
    const float* xr = x + r * D;
    float* yr = y + r * D;
    float s = 0.0f;
    for (int c = threadIdx.x; c < D; c += 128) s += xr[c];
    red[threadIdx.x] = s; __syncthreads();
    for (int o = 64; o > 0; o >>= 1) {
        if (threadIdx.x < o) red[threadIdx.x] += red[threadIdx.x + o];
        __syncthreads();
    }
    const float mean = red[0] / D; __syncthreads();
    float v = 0.0f;
    for (int c = threadIdx.x; c < D; c += 128) { float d = xr[c] - mean; v += d * d; }
    red[threadIdx.x] = v; __syncthreads();
    for (int o = 64; o > 0; o >>= 1) {
        if (threadIdx.x < o) red[threadIdx.x] += red[threadIdx.x + o];
        __syncthreads();
    }
    const float rstd = rsqrtf(red[0] / D + 1e-5f);
    for (int c = threadIdx.x; c < D; c += 128)
        yr[c] = (xr[c] - mean) * rstd * g[c] + b[c];
}

// row-wise softmax in place
__global__ void __launch_bounds__(256)
softmax_kernel(float* __restrict__ x, int L)
{
    const long r = blockIdx.x;
    float* xr = x + r * (long)L;
    __shared__ float red[256];
    float mx = -1e30f;
    for (int c = threadIdx.x; c < L; c += 256) mx = fmaxf(mx, xr[c]);
    red[threadIdx.x] = mx; __syncthreads();
    for (int o = 128; o > 0; o >>= 1) {
        if (threadIdx.x < o) red[threadIdx.x] = fmaxf(red[threadIdx.x], red[threadIdx.x + o]);
        __syncthreads();
    }
    mx = red[0]; __syncthreads();
    float s = 0.0f;
    for (int c = threadIdx.x; c < L; c += 256) { float e = __expf(xr[c] - mx); xr[c] = e; s += e; }
    red[threadIdx.x] = s; __syncthreads();
    for (int o = 128; o > 0; o >>= 1) {
        if (threadIdx.x < o) red[threadIdx.x] += red[threadIdx.x + o];
        __syncthreads();
    }
    const float inv = 1.0f / red[0];
    for (int c = threadIdx.x; c < L; c += 256) xr[c] *= inv;
}

// feat[b] = [mean_g x, max_g x]
__global__ void __launch_bounds__(256)
meanmax_kernel(const float* __restrict__ x, float* __restrict__ feat, int G, int D)
{
    const int b = blockIdx.x;
    const float* xb = x + (long)b * G * D;
    for (int c = threadIdx.x; c < D; c += 256) {
        float s = 0.0f, m = -1e30f;
        for (int g = 0; g < G; ++g) { float v = xb[(long)g * D + c]; s += v; m = fmaxf(m, v); }
        feat[(long)b * 2 * D + c]     = s / G;
        feat[(long)b * 2 * D + D + c] = m;
    }
}

// row-wise log_softmax in place
__global__ void __launch_bounds__(256)
log_softmax_kernel(float* __restrict__ x, int L)
{
    const int b = blockIdx.x;
    float* xr = x + (long)b * L;
    __shared__ float red[256];
    float mx = -1e30f;
    for (int c = threadIdx.x; c < L; c += 256) mx = fmaxf(mx, xr[c]);
    red[threadIdx.x] = mx; __syncthreads();
    for (int o = 128; o > 0; o >>= 1) {
        if (threadIdx.x < o) red[threadIdx.x] = fmaxf(red[threadIdx.x], red[threadIdx.x + o]);
        __syncthreads();
    }
    mx = red[0]; __syncthreads();
    float s = 0.0f;
    for (int c = threadIdx.x; c < L; c += 256) s += __expf(xr[c] - mx);
    red[threadIdx.x] = s; __syncthreads();
    for (int o = 128; o > 0; o >>= 1) {
        if (threadIdx.x < o) red[threadIdx.x] += red[threadIdx.x + o];
        __syncthreads();
    }
    const float lse = mx + __logf(red[0]);
    for (int c = threadIdx.x; c < L; c += 256) xr[c] -= lse;
}

// ---------------------------------------------------------------------------
// Host-side helpers
// ---------------------------------------------------------------------------
static inline unsigned int nblk(long n, int bs) { return (unsigned int)((n + bs - 1) / bs); }

static void run_gemm(hipStream_t stream, int epi, int transB,
                     const float* A, int lda, long bsA, long hsA,
                     const float* B, int ldb, long bsB, long hsB,
                     const float* bias,
                     const float* g, const float* bb, const float* m, const float* v,
                     float* C, int ldc, long bsC, long hsC,
                     int M, int N, int K, int Z, int H, float alpha)
{
    dim3 grid((unsigned)((M + 127) / 128), (unsigned)((N + 63) / 64), (unsigned)Z);
#define GEMM_CALL(E, T)                                                          \
    gemm_wmma<E, T><<<grid, dim3(256), 0, stream>>>(                             \
        A, lda, bsA, hsA, B, ldb, bsB, hsB, bias, g, bb, m, v,                   \
        C, ldc, bsC, hsC, M, N, K, H, alpha)
    if (transB) {
        if (epi == 0) GEMM_CALL(0, 1);
        else if (epi == 1) GEMM_CALL(1, 1);
        else if (epi == 2) GEMM_CALL(2, 1);
        else GEMM_CALL(3, 1);
    } else {
        if (epi == 0) GEMM_CALL(0, 0);
        else if (epi == 1) GEMM_CALL(1, 0);
        else if (epi == 2) GEMM_CALL(2, 0);
        else GEMM_CALL(3, 0);
    }
#undef GEMM_CALL
}

// ---------------------------------------------------------------------------
// kernel_launch: full Point-M2AE-ReID forward pass
// Input leaf order assumes jax tree flattening (alphabetical dict keys):
//   params.blocks (3 lvls x 5 blks x 11 leaves) -> 0..164
//   enc_norm_b 165, enc_norm_g 166, final_norm_b 167, final_norm_g 168
//   head: b1 169, b2 170, b3 171, bn1{b,g,m,v} 172-175, bn2{b,g,m,v} 176-179,
//         w1 180, w2 181, w3 182
//   pos_embed lvl i: base 183+4i -> b1,b2,w1,w2
//   token_embed lvl i: base 195+16i -> b1,b2,b3,b4,bn1{b,g,m,v},bn2{b,g,m,v},
//                                      w1,w2,w3,w4
//   pts 243
// ---------------------------------------------------------------------------
extern "C" void kernel_launch(void* const* d_in, const int* in_sizes, int n_in,
                              void* d_out, int out_size, void* d_ws, size_t ws_size,
                              hipStream_t stream)
{
    (void)in_sizes; (void)n_in; (void)out_size; (void)ws_size;

    const int BATCH = 32, NPTS = 2048, HEADS = 6;
    const int NG[3]  = {512, 256, 64};
    const int KS[3]  = {16, 8, 8};
    const int DIM[3] = {96, 192, 384};

    auto P = [&](int i) -> const float* { return (const float*)d_in[i]; };
    // block leaf: 0 fc1_b,1 fc1_w,2 fc2_b,3 fc2_w,4 ln1b,5 ln1g,6 ln2b,7 ln2g,
    //             8 proj_b,9 proj_w,10 qkv_w
    auto BP = [&](int lvl, int j, int leaf) -> const float* {
        return P((lvl * 5 + j) * 11 + leaf);
    };
    const float* pts = P(243);

    // Workspace bump allocator (256B aligned)
    char* wsp = (char*)d_ws;
    size_t off = 0;
    auto alloc_f = [&](size_t n) -> float* {
        float* p = (float*)(wsp + off);
        off += ((n * sizeof(float) + 255) & ~(size_t)255);
        return p;
    };
    auto alloc_i = [&](size_t n) -> int* {
        int* p = (int*)(wsp + off);
        off += ((n * sizeof(int) + 255) & ~(size_t)255);
        return p;
    };

    int*   cidx0 = alloc_i((size_t)BATCH * 512);
    float* ctr0  = alloc_f((size_t)BATCH * 512 * 3);
    int*   cidx1 = alloc_i((size_t)BATCH * 256);
    float* ctr1  = alloc_f((size_t)BATCH * 256 * 3);
    int*   cidx2 = alloc_i((size_t)BATCH * 64);
    float* ctr2  = alloc_f((size_t)BATCH * 64 * 3);
    int*   nidx0 = alloc_i((size_t)BATCH * 512 * 16);
    int*   nidx1 = alloc_i((size_t)BATCH * 256 * 8);
    int*   nidx2 = alloc_i((size_t)BATCH * 64 * 8);
    float* nb0   = alloc_f((size_t)BATCH * 512 * 16 * 3);
    float* inpbuf= alloc_f((size_t)BATCH * 256 * 8 * 96);   // max gathered feats
    float* f1    = alloc_f((size_t)262144 * 48);            // max RK*h
    float* f2    = alloc_f((size_t)262144 * 48);
    float* catb  = alloc_f((size_t)262144 * 96);            // max RK*D
    float* f3    = alloc_f((size_t)262144 * 96);
    float* f4    = alloc_f((size_t)262144 * 96);
    float* token = alloc_f((size_t)BATCH * 49152);          // max G*D
    float* pos   = alloc_f((size_t)BATCH * 49152);
    float* tmpP  = alloc_f((size_t)BATCH * 49152);
    float* x     = alloc_f((size_t)BATCH * 49152);
    float* xin   = alloc_f((size_t)BATCH * 49152);
    float* tbuf  = alloc_f((size_t)BATCH * 49152);
    float* qkv   = alloc_f((size_t)BATCH * 512 * 288);      // max G*3D
    float* scores= alloc_f((size_t)BATCH * HEADS * 512 * 512);
    float* obuf  = alloc_f((size_t)BATCH * 49152);
    float* m1    = alloc_f((size_t)BATCH * 196608);         // max G*4D
    float* m2    = alloc_f((size_t)BATCH * 49152);
    float* feat  = alloc_f((size_t)BATCH * 768);
    float* h1    = alloc_f((size_t)BATCH * 256);
    float* h2    = alloc_f((size_t)BATCH * 256);

    // ---- Grouping: FPS + centers + KNN + level-0 neighborhoods ----
    fps_kernel<<<BATCH, 256, 0, stream>>>(pts, NPTS, NG[0], cidx0);
    { long tot = (long)BATCH * NG[0] * 3;
      gather_points<<<nblk(tot, 256), 256, 0, stream>>>(pts, cidx0, ctr0, NPTS, NG[0], 3, tot); }
    knn_kernel<16><<<nblk((long)BATCH * NG[0], 128), 128, 0, stream>>>(
        pts, ctr0, NPTS, NG[0], BATCH * NG[0], nidx0);
    { long tot = (long)BATCH * NG[0] * KS[0] * 3;
      gather_sub<<<nblk(tot, 256), 256, 0, stream>>>(pts, nidx0, ctr0, nb0, NPTS, NG[0], KS[0], tot); }

    fps_kernel<<<BATCH, 256, 0, stream>>>(ctr0, NG[0], NG[1], cidx1);
    { long tot = (long)BATCH * NG[1] * 3;
      gather_points<<<nblk(tot, 256), 256, 0, stream>>>(ctr0, cidx1, ctr1, NG[0], NG[1], 3, tot); }
    knn_kernel<8><<<nblk((long)BATCH * NG[1], 128), 128, 0, stream>>>(
        ctr0, ctr1, NG[0], NG[1], BATCH * NG[1], nidx1);

    fps_kernel<<<BATCH, 256, 0, stream>>>(ctr1, NG[1], NG[2], cidx2);
    { long tot = (long)BATCH * NG[2] * 3;
      gather_points<<<nblk(tot, 256), 256, 0, stream>>>(ctr1, cidx2, ctr2, NG[1], NG[2], 3, tot); }
    knn_kernel<8><<<nblk((long)BATCH * NG[2], 128), 128, 0, stream>>>(
        ctr1, ctr2, NG[1], NG[2], BATCH * NG[2], nidx2);

    // ---- Hierarchical encoder ----
    int Gprev = 0, Dprev = 0;
    for (int i = 0; i < 3; ++i) {
        const int G = NG[i], K = KS[i], D = DIM[i], hh = D / 2;
        const int ic = (i == 0) ? 3 : DIM[i - 1];
        const long R = (long)BATCH * G, RK = R * K;
        const float* ctr  = (i == 0) ? ctr0 : (i == 1 ? ctr1 : ctr2);
        const int*   nidx = (i == 0) ? nidx0 : (i == 1 ? nidx1 : nidx2);

        // token embed input
        const float* inp;
        if (i == 0) inp = nb0;
        else {
            long tot = RK * ic;
            gather_feats<<<nblk(tot, 256), 256, 0, stream>>>(
                x, nidx, inpbuf, Gprev, Dprev, G, K, tot);
            inp = inpbuf;
        }

        // mini-PointNet
        const int tb = 195 + 16 * i;
        run_gemm(stream, 3, 1, inp, ic, 0, 0, P(tb + 12), ic, 0, 0, P(tb + 0),
                 P(tb + 5), P(tb + 4), P(tb + 6), P(tb + 7),
                 f1, hh, 0, 0, (int)RK, hh, ic, 1, 1, 1.0f);
        run_gemm(stream, 0, 1, f1, hh, 0, 0, P(tb + 13), hh, 0, 0, P(tb + 1),
                 nullptr, nullptr, nullptr, nullptr,
                 f2, hh, 0, 0, (int)RK, hh, hh, 1, 1, 1.0f);
        concat_maxpool<<<(unsigned)R, 128, 0, stream>>>(f2, catb, K, hh);
        run_gemm(stream, 3, 1, catb, D, 0, 0, P(tb + 14), D, 0, 0, P(tb + 2),
                 P(tb + 9), P(tb + 8), P(tb + 10), P(tb + 11),
                 f3, D, 0, 0, (int)RK, D, D, 1, 1, 1.0f);
        run_gemm(stream, 0, 1, f3, D, 0, 0, P(tb + 15), D, 0, 0, P(tb + 3),
                 nullptr, nullptr, nullptr, nullptr,
                 f4, D, 0, 0, (int)RK, D, D, 1, 1, 1.0f);
        maxpool_k<<<nblk(R * D, 256), 256, 0, stream>>>(f4, token, K, D, R * D);

        // pos embed
        const int pb = 183 + 4 * i;   // b1,b2,w1,w2
        run_gemm(stream, 2, 1, ctr, 3, 0, 0, P(pb + 2), 3, 0, 0, P(pb + 0),
                 nullptr, nullptr, nullptr, nullptr,
                 tmpP, D, 0, 0, (int)R, D, 3, 1, 1, 1.0f);
        run_gemm(stream, 0, 1, tmpP, D, 0, 0, P(pb + 3), D, 0, 0, P(pb + 1),
                 nullptr, nullptr, nullptr, nullptr,
                 pos, D, 0, 0, (int)R, D, D, 1, 1, 1.0f);

        // transformer blocks
        const int dh = D / HEADS;
        const float scale = 1.0f / sqrtf((float)dh);
        for (int j = 0; j < 5; ++j) {
            const float* src = (j == 0) ? token : x;
            add_kernel<<<nblk(R * D, 256), 256, 0, stream>>>(src, pos, xin, R * D);
            layernorm_kernel<<<(unsigned)R, 128, 0, stream>>>(
                xin, BP(i, j, 5), BP(i, j, 4), tbuf, D);
            run_gemm(stream, 0, 1, tbuf, D, 0, 0, BP(i, j, 10), D, 0, 0, nullptr,
                     nullptr, nullptr, nullptr, nullptr,
                     qkv, 3 * D, 0, 0, (int)R, 3 * D, D, 1, 1, 1.0f);
            // scores = softmax(Q K^T * dh^-0.5)   [batched over (b,h)]
            run_gemm(stream, 0, 1,
                     qkv, 3 * D, (long)G * 3 * D, (long)dh,
                     qkv + D, 3 * D, (long)G * 3 * D, (long)dh,
                     nullptr, nullptr, nullptr, nullptr, nullptr,
                     scores, G, (long)HEADS * G * G, (long)G * G,
                     G, G, dh, BATCH * HEADS, HEADS, scale);
            softmax_kernel<<<(unsigned)((long)BATCH * HEADS * G), 256, 0, stream>>>(scores, G);
            // O = P V
            run_gemm(stream, 0, 0,
                     scores, G, (long)HEADS * G * G, (long)G * G,
                     qkv + 2 * D, 3 * D, (long)G * 3 * D, (long)dh,
                     nullptr, nullptr, nullptr, nullptr, nullptr,
                     obuf, D, (long)G * D, (long)dh,
                     G, dh, G, BATCH * HEADS, HEADS, 1.0f);
            run_gemm(stream, 0, 1, obuf, D, 0, 0, BP(i, j, 9), D, 0, 0, BP(i, j, 8),
                     nullptr, nullptr, nullptr, nullptr,
                     m2, D, 0, 0, (int)R, D, D, 1, 1, 1.0f);
            add_kernel<<<nblk(R * D, 256), 256, 0, stream>>>(xin, m2, x, R * D);
            // MLP
            layernorm_kernel<<<(unsigned)R, 128, 0, stream>>>(
                x, BP(i, j, 7), BP(i, j, 6), tbuf, D);
            run_gemm(stream, 2, 1, tbuf, D, 0, 0, BP(i, j, 1), D, 0, 0, BP(i, j, 0),
                     nullptr, nullptr, nullptr, nullptr,
                     m1, 4 * D, 0, 0, (int)R, 4 * D, D, 1, 1, 1.0f);
            run_gemm(stream, 0, 1, m1, 4 * D, 0, 0, BP(i, j, 3), 4 * D, 0, 0, BP(i, j, 2),
                     nullptr, nullptr, nullptr, nullptr,
                     m2, D, 0, 0, (int)R, D, 4 * D, 1, 1, 1.0f);
            add_kernel<<<nblk(R * D, 256), 256, 0, stream>>>(x, m2, x, R * D);
        }
        Gprev = G; Dprev = D;
    }

    // ---- Final norms + pooled features + head ----
    const int Gf = NG[2], Df = DIM[2];
    const long Rf = (long)BATCH * Gf;
    layernorm_kernel<<<(unsigned)Rf, 128, 0, stream>>>(x, P(166), P(165), tbuf, Df);
    layernorm_kernel<<<(unsigned)Rf, 128, 0, stream>>>(tbuf, P(168), P(167), xin, Df);
    meanmax_kernel<<<BATCH, 256, 0, stream>>>(xin, feat, Gf, Df);

    run_gemm(stream, 3, 1, feat, 2 * Df, 0, 0, P(180), 2 * Df, 0, 0, P(169),
             P(173), P(172), P(174), P(175),
             h1, 256, 0, 0, BATCH, 256, 2 * Df, 1, 1, 1.0f);
    run_gemm(stream, 3, 1, h1, 256, 0, 0, P(181), 256, 0, 0, P(170),
             P(177), P(176), P(178), P(179),
             h2, 256, 0, 0, BATCH, 256, 256, 1, 1, 1.0f);
    run_gemm(stream, 0, 1, h2, 256, 0, 0, P(182), 256, 0, 0, P(171),
             nullptr, nullptr, nullptr, nullptr,
             (float*)d_out, 751, 0, 0, BATCH, 751, 256, 1, 1, 1.0f);
    log_softmax_kernel<<<BATCH, 256, 0, stream>>>((float*)d_out, 751);
}
